// FullAttention_37598143709779
// MI455X (gfx1250) — matrix operand order repaired and checked by reading, not compiled, not run on hardware
//
#include <hip/hip_runtime.h>

// ---------------- problem constants ----------------
constexpr int BB = 4;
constexpr int NN = 2048;
constexpr int CC = 512;
constexpr int HH = 8;
constexpr int HD = 64;          // head dim
constexpr int M_ROWS = BB * NN; // 8192 GEMM rows

typedef __attribute__((ext_vector_type(16))) _Float16 v16h;
typedef __attribute__((ext_vector_type(8)))  float    v8f;
typedef __attribute__((ext_vector_type(4)))  unsigned v4u;
typedef __attribute__((ext_vector_type(8)))  unsigned v8u;

#define DEV __device__ __forceinline__

// ---------------- WMMA helpers ----------------
DEV v8f wmma_f16(v16h a, v16h b, v8f c) {
  // D = A(16x32 f16) * B(32x16 f16) + C(16x16 f32)
  return __builtin_amdgcn_wmma_f32_16x16x32_f16(
      /*neg_a=*/false, a, /*neg_b=*/false, b,
      /*c_mod=*/(short)0, c, /*reuse_a=*/false, /*reuse_b=*/false);
}

// A fragment: 16x32 f16 tile from row-major matrix p (leading dim ld, halfs).
// Contiguous-k layout used here (same as B): lane L holds row (L&15); VGPR v holds K = 2v + 16*(L>>4), +1.
DEV v16h load_frag_a(const _Float16* __restrict__ p, int ld, int row0, int k0) {
  const int lane = threadIdx.x & 31;
  const int m  = lane & 15;
  const int hh = (lane >> 4) & 1;
  const _Float16* base = p + (size_t)(row0 + m) * ld + k0 + hh * 16;
  v16h a;
#pragma unroll
  for (int v = 0; v < 8; ++v) {
    const int k = 2 * v;
    a[2 * v]     = base[k];
    a[2 * v + 1] = base[k + 1];
  }
  return a;
}

// B fragment: 32x16 f16 where B[k][n] = p[n0+n][k0+k] (weight-style row-major,
// rows indexed by the WMMA "n" dim). Contiguous-k layout used here (as in A): lane L holds col (L&15);
// VGPR v holds K = 2v + 16*(L>>4), +1  -> 16 contiguous halfs per lane.
DEV v16h load_frag_b(const _Float16* __restrict__ p, int ld, int n0, int k0) {
  const int lane = threadIdx.x & 31;
  const int n  = lane & 15;
  const int hh = (lane >> 4) & 1;
  const _Float16* base = p + (size_t)(n0 + n) * ld + k0 + hh * 16;
  v16h b;
#pragma unroll
  for (int e = 0; e < 16; ++e) b[e] = base[e];
  return b;
}

// Same as load_frag_b but for LDS-resident tiles (ds_load path after
// address-space inference; callers pass pointers into __shared__ arrays).
DEV v16h load_frag_b_lds(const _Float16* p, int ld, int n0, int k0) {
  const int lane = threadIdx.x & 31;
  const int n  = lane & 15;
  const int hh = (lane >> 4) & 1;
  const _Float16* base = p + (n0 + n) * ld + k0 + hh * 16;
  v16h b;
#pragma unroll
  for (int e = 0; e < 16; ++e) b[e] = base[e];
  return b;
}

// ---------------- Tensor Data Mover: async 2D tile -> LDS ----------------
// Builds a 2-group D# (2D tensor) per CDNA5 ISA 8.3/8.4 and issues
// tensor_load_to_lds (VADDR2/3 disabled). Tracked by TENSORcnt.
// dims/strides/tiles in elements (data_size = 2 bytes). LDS pad feature used
// to insert pad_amount DWORDs after each pad_interval DWORDs.
DEV void tdm_load_2d_f16(unsigned lds_byte_addr, const _Float16* gptr,
                         unsigned dim0, unsigned dim1, unsigned stride0,
                         unsigned tile0, unsigned tile1,
                         unsigned pad_interval_code, unsigned pad_amount_code) {
  const unsigned long long ga = (unsigned long long)(size_t)gptr;
  v4u g0;
  g0[0] = 1u;                                        // count=1, user mode
  g0[1] = lds_byte_addr;                             // lds_addr
  g0[2] = (unsigned)(ga & 0xffffffffu);              // global_addr[31:0]
  g0[3] = (unsigned)((ga >> 32) & 0x1ffffffu)        // global_addr[56:32]
          | (2u << 30);                              // type = 2 (image)
  v8u g1;
  g1[0] = (1u << 16)                                 // data_size = 2 bytes
        | (1u << 20)                                 // pad_enable
        | (pad_interval_code << 22)
        | (pad_amount_code << 25);
  g1[1] = (dim0 & 0xffffu) << 16;                    // tensor_dim0[15:0]
  g1[2] = (dim0 >> 16) | ((dim1 & 0xffffu) << 16);   // dim0[31:16] | dim1[15:0]
  g1[3] = (dim1 >> 16) | (tile0 << 16);              // dim1[31:16] | tile_dim0
  g1[4] = tile1 & 0xffffu;                           // tile_dim1 (tile_dim2=0)
  g1[5] = stride0;                                   // tensor_dim0_stride[31:0]
  g1[6] = 0;                                         // stride0 hi / dim1_stride lo
  g1[7] = 0;
  asm volatile("tensor_load_to_lds %0, %1" :: "s"(g0), "s"(g1) : "memory");
}

// ---------------- kernel 1: fp32 -> f16 convert ----------------
__global__ void convert_f16_kernel(const float* __restrict__ in,
                                   _Float16* __restrict__ out, int n) {
  for (int i = blockIdx.x * blockDim.x + threadIdx.x; i < n;
       i += gridDim.x * blockDim.x)
    out[i] = (_Float16)in[i];
}

// ---------------- kernel 2: QKV projection ----------------
// qkv[m][d] = sum_c xh[m][c] * wq[d][c]      (m = b*N + t, d in [0,3C))
// Scatter: d<512 -> Q(B,H,N,HD); d<1024 -> K(B,H,N,HD); else -> Vt(B,H,HD,N)
__global__ __launch_bounds__(256) void qkv_gemm_kernel(
    const _Float16* __restrict__ xh, const _Float16* __restrict__ wq,
    _Float16* __restrict__ q, _Float16* __restrict__ kmat,
    _Float16* __restrict__ vt) {
  const int wave = threadIdx.x >> 5;
  const int lane = threadIdx.x & 31;
  const int m0 = blockIdx.x * 128 + wave * 16;
  const int n0 = blockIdx.y * 64;

  v8f acc[4] = {v8f{0}, v8f{0}, v8f{0}, v8f{0}};
#pragma unroll 2
  for (int k0 = 0; k0 < CC; k0 += 32) {
    // speculative prefetch of the A row two tiles ahead (branch-free; a
    // past-the-end prefetch is silently dropped by hardware)
    __builtin_prefetch(xh + (size_t)(m0 + (lane & 15)) * CC + k0 + 64, 0, 1);
    const v16h a = load_frag_a(xh, CC, m0, k0);
#pragma unroll
    for (int f = 0; f < 4; ++f)
      acc[f] = wmma_f16(a, load_frag_b(wq, CC, n0 + f * 16, k0), acc[f]);
  }
#pragma unroll
  for (int f = 0; f < 4; ++f) {
    const int d = n0 + f * 16 + (lane & 15);
    const int which = d >> 9;      // 0:Q 1:K 2:V
    const int dc = d & 511;
    const int h = dc >> 6, hd = dc & 63;
#pragma unroll
    for (int v = 0; v < 8; ++v) {
      const int m = m0 + v + ((lane >> 4) << 3);
      const int b = m >> 11, t = m & (NN - 1);
      const _Float16 val = (_Float16)acc[f][v];
      const size_t bh = (size_t)b * HH + h;
      if (which == 0)      q[(bh * NN + t) * HD + hd] = val;
      else if (which == 1) kmat[(bh * NN + t) * HD + hd] = val;
      else                 vt[(bh * HD + hd) * NN + t] = val;   // transposed
    }
  }
}

// ---------------- kernel 3: flash attention ----------------
// One block = 4 waves = 64 query rows of one (b,h). Keys streamed 32 at a
// time; K/V tiles staged in LDS by the Tensor Data Mover (double-buffered,
// wave 0 issues, TENSORcnt + barrier hand-off). TDM pads LDS rows:
//   K tile 32x64h  -> row stride 72h (interval 32 dw, pad 4 dw)
//   V tile 64x32h  -> row stride 40h (interval 16 dw, pad 4 dw)
__global__ __launch_bounds__(128) void flash_attn_kernel(
    const _Float16* __restrict__ q, const _Float16* __restrict__ kmat,
    const _Float16* __restrict__ vt, _Float16* __restrict__ ao) {
  __shared__ _Float16 ldsK[2][32][72];
  __shared__ _Float16 ldsV[2][64][40];
  __shared__ _Float16 ldsP[4][16][40];   // per-wave 16x32 P tile

  const int wave = threadIdx.x >> 5;
  const int lane = threadIdx.x & 31;
  const int bh = blockIdx.x;
  const int b = bh >> 3, h = bh & 7;
  const int q0 = blockIdx.y * 64 + wave * 16;

  const _Float16* qp = q    + (size_t)bh * NN * HD;
  const _Float16* kp = kmat + (size_t)bh * NN * HD;
  const _Float16* vp = vt   + (size_t)bh * HD * NN;

  const v16h aq0 = load_frag_a(qp, HD, q0, 0);
  const v16h aq1 = load_frag_a(qp, HD, q0, 32);

  // prologue: stage first K/V tile into buffer 0
  if (wave == 0) {
    tdm_load_2d_f16((unsigned)(size_t)&ldsK[0][0][0], kp,
                    HD, NN, HD, /*tile*/ HD, 32, /*pad*/ 4, 3);
    tdm_load_2d_f16((unsigned)(size_t)&ldsV[0][0][0], vp,
                    NN, HD, NN, /*tile*/ 32, HD, /*pad*/ 3, 3);
  }

  v8f o[4] = {v8f{0}, v8f{0}, v8f{0}, v8f{0}};
  float mx[8], l[8];
#pragma unroll
  for (int v = 0; v < 8; ++v) { mx[v] = -1.0e30f; l[v] = 0.0f; }
  const float scale = 0.125f;  // 1/sqrt(64)

  for (int j = 0; j < NN; j += 32) {
    const int cur = (j >> 5) & 1;
    if (wave == 0) {
      if (j + 32 < NN) {  // kick DMA for next tile, then wait for current pair
        tdm_load_2d_f16((unsigned)(size_t)&ldsK[cur ^ 1][0][0],
                        kp + (size_t)(j + 32) * HD,
                        HD, NN, HD, HD, 32, 4, 3);
        tdm_load_2d_f16((unsigned)(size_t)&ldsV[cur ^ 1][0][0],
                        vp + (j + 32),
                        NN, HD, NN, 32, HD, 3, 3);
        __builtin_amdgcn_s_wait_tensorcnt(2);  // in-order: current pair done
      } else {
        __builtin_amdgcn_s_wait_tensorcnt(0);
      }
    }
    __syncthreads();  // current buffer visible to all waves

    // S tiles from LDS K: B[k=d][n=key] = ldsK[key][d]
    const _Float16* kt = &ldsK[cur][0][0];
    v8f s0 = v8f{0}, s1 = v8f{0};
    s0 = wmma_f16(aq0, load_frag_b_lds(kt, 72, 0, 0), s0);
    s0 = wmma_f16(aq1, load_frag_b_lds(kt, 72, 0, 32), s0);
    s1 = wmma_f16(aq0, load_frag_b_lds(kt, 72, 16, 0), s1);
    s1 = wmma_f16(aq1, load_frag_b_lds(kt, 72, 16, 32), s1);

    float alpha[8];
#pragma unroll
    for (int v = 0; v < 8; ++v) {
      const float a0 = s0[v] * scale, a1 = s1[v] * scale;
      float mnew = fmaxf(a0, a1);
#pragma unroll
      for (int off = 1; off < 16; off <<= 1)   // row lives in one 16-lane half
        mnew = fmaxf(mnew, __shfl_xor(mnew, off, 16));
      mnew = fmaxf(mnew, mx[v]);
      alpha[v] = __expf(mx[v] - mnew);
      const float p0 = __expf(a0 - mnew), p1 = __expf(a1 - mnew);
      float rs = p0 + p1;
#pragma unroll
      for (int off = 1; off < 16; off <<= 1)
        rs += __shfl_xor(rs, off, 16);
      l[v] = l[v] * alpha[v] + rs;
      mx[v] = mnew;
      const int r = v + ((lane >> 4) << 3);    // C-layout row
      ldsP[wave][r][lane & 15]        = (_Float16)p0;
      ldsP[wave][r][16 + (lane & 15)] = (_Float16)p1;
    }
    // wave-private transpose: per-wave LDS ops are in-order, no barrier needed
    v16h ap;
    {
      const int m = lane & 15, hh = (lane >> 4) & 1;
#pragma unroll
      for (int v = 0; v < 8; ++v) {
        const int k = 2 * v + hh * 16;
        ap[2 * v]     = ldsP[wave][m][k];
        ap[2 * v + 1] = ldsP[wave][m][k + 1];
      }
    }

    // online rescale + O += P @ V   (B[k=key][n=d] = ldsV[d][key])
#pragma unroll
    for (int v = 0; v < 8; ++v) {
      o[0][v] *= alpha[v]; o[1][v] *= alpha[v];
      o[2][v] *= alpha[v]; o[3][v] *= alpha[v];
    }
    const _Float16* vtile = &ldsV[cur][0][0];
#pragma unroll
    for (int f = 0; f < 4; ++f)
      o[f] = wmma_f16(ap, load_frag_b_lds(vtile, 40, f * 16, 0), o[f]);

    __syncthreads();  // all waves done with cur before TDM overwrites it
  }

  // normalize and store to (B,N,C) f16:  c = h*HD + f*16 + (lane&15)
#pragma unroll
  for (int f = 0; f < 4; ++f) {
    const int c = h * HD + f * 16 + (lane & 15);
#pragma unroll
    for (int v = 0; v < 8; ++v) {
      const int t = q0 + v + ((lane >> 4) << 3);
      ao[((size_t)b * NN + t) * CC + c] = (_Float16)(o[f][v] / l[v]);
    }
  }
}

// ---------------- kernel 4: output projection ----------------
// out[m][d] = sum_c ao[m][c] * wp[d][c] + bias[d]   (fp32 out)
__global__ __launch_bounds__(256) void proj_gemm_kernel(
    const _Float16* __restrict__ ao, const _Float16* __restrict__ wp,
    const float* __restrict__ bias, float* __restrict__ out) {
  const int wave = threadIdx.x >> 5;
  const int lane = threadIdx.x & 31;
  const int m0 = blockIdx.x * 128 + wave * 16;
  const int n0 = blockIdx.y * 64;

  v8f acc[4] = {v8f{0}, v8f{0}, v8f{0}, v8f{0}};
#pragma unroll 2
  for (int k0 = 0; k0 < CC; k0 += 32) {
    __builtin_prefetch(ao + (size_t)(m0 + (lane & 15)) * CC + k0 + 64, 0, 1);
    const v16h a = load_frag_a(ao, CC, m0, k0);
#pragma unroll
    for (int f = 0; f < 4; ++f)
      acc[f] = wmma_f16(a, load_frag_b(wp, CC, n0 + f * 16, k0), acc[f]);
  }
#pragma unroll
  for (int f = 0; f < 4; ++f) {
    const int d = n0 + f * 16 + (lane & 15);
    const float bval = bias[d];
#pragma unroll
    for (int v = 0; v < 8; ++v) {
      const int m = m0 + v + ((lane >> 4) << 3);
      out[(size_t)m * CC + d] = acc[f][v] + bval;
    }
  }
}

// ---------------- host launcher ----------------
extern "C" void kernel_launch(void* const* d_in, const int* in_sizes, int n_in,
                              void* d_out, int out_size, void* d_ws,
                              size_t ws_size, hipStream_t stream) {
  const float* x      = (const float*)d_in[0];
  const float* w_qkv  = (const float*)d_in[1];
  const float* w_proj = (const float*)d_in[2];
  const float* b_proj = (const float*)d_in[3];
  float* out = (float*)d_out;

  char* ws = (char*)d_ws;
  size_t off = 0;
  auto alloc = [&](size_t bytes) -> void* {
    void* p = ws + off;
    off += (bytes + 255) & ~(size_t)255;
    return p;
  };

  _Float16* xh   = (_Float16*)alloc((size_t)M_ROWS * CC * 2);       // 8 MB
  _Float16* wqh  = (_Float16*)alloc((size_t)3 * CC * CC * 2);       // 1.5 MB
  _Float16* wph  = (_Float16*)alloc((size_t)CC * CC * 2);           // 0.5 MB
  _Float16* qbuf = (_Float16*)alloc((size_t)BB * HH * NN * HD * 2); // 8 MB
  _Float16* kbuf = (_Float16*)alloc((size_t)BB * HH * NN * HD * 2); // 8 MB
  _Float16* vtb  = (_Float16*)alloc((size_t)BB * HH * HD * NN * 2); // 8 MB
  _Float16* ao   = (_Float16*)alloc((size_t)M_ROWS * CC * 2);       // 8 MB

  // 1) converts
  convert_f16_kernel<<<2048, 256, 0, stream>>>(x, xh, M_ROWS * CC);
  convert_f16_kernel<<<768, 256, 0, stream>>>(w_qkv, wqh, 3 * CC * CC);
  convert_f16_kernel<<<256, 256, 0, stream>>>(w_proj, wph, CC * CC);

  // 2) QKV projection: 8192 x 1536 x 512
  dim3 g1(M_ROWS / 128, (3 * CC) / 64);
  qkv_gemm_kernel<<<g1, 256, 0, stream>>>(xh, wqh, qbuf, kbuf, vtb);

  // 3) flash attention: one block per (b*h, 64-query tile)
  dim3 g2(BB * HH, NN / 64);
  flash_attn_kernel<<<g2, 128, 0, stream>>>(qbuf, kbuf, vtb, ao);

  // 4) output projection: 8192 x 512 x 512 (+bias, fp32 out)
  dim3 g3(M_ROWS / 128, CC / 64);
  proj_gemm_kernel<<<g3, 256, 0, stream>>>(ao, wph, b_proj, out);
}
